// Attention_3418793967804
// MI455X (gfx1250) — compile-verified, matrix operand order checked by reading, not run on hardware
//
#include <hip/hip_runtime.h>
#include <hip/hip_bf16.h>
#include <stdint.h>

// ---------------------------------------------------------------------------
// Types for CDNA5 WMMA (wave32): D(16x16 f32) = A(16x32 bf16) x B(32x16 bf16) + C
// ---------------------------------------------------------------------------
typedef __attribute__((ext_vector_type(16))) __bf16        v16bf;
typedef __attribute__((ext_vector_type(8)))  float         v8f;
typedef __attribute__((ext_vector_type(4)))  unsigned int  v4u;

union Frag { v16bf v; v4u u[2]; __bf16 h[16]; };

static __device__ __forceinline__ __bf16 f2bf(float f) {
  union { float f; uint32_t u; } a; a.f = f;
  uint32_t r = a.u + 0x7FFFu + ((a.u >> 16) & 1u);   // round-to-nearest-even
  union { unsigned short s; __bf16 h; } o; o.s = (unsigned short)(r >> 16);
  return o.h;
}
static __device__ __forceinline__ float bf2f(__bf16 h) {
  union { unsigned short s; __bf16 h; } i; i.h = h;
  union { uint32_t u; float f; } o; o.u = ((uint32_t)i.s) << 16;
  return o.f;
}

// A/B fragment (16-bit, 16x32 / 32x16): per lane, elements 0..7 come from the
// contiguous K-run at (half*8), elements 8..15 from the run at (16 + half*8).
static __device__ __forceinline__ v16bf load_frag(const __bf16* lo, const __bf16* hi) {
  Frag f;
  f.u[0] = *(const v4u*)lo;
  f.u[1] = *(const v4u*)hi;
  return f.v;
}

#define WMMA_BF16(A, B, C) \
  __builtin_amdgcn_wmma_f32_16x16x32_bf16(false, (A), false, (B), (short)0, (C), false, false)

// Problem constants
#define BATCH   8
#define CH      256
#define HHW     48
#define HW      2304          // 48*48
#define NTOK    18432         // 8*2304
#define NHEAD   4
#define DK      32
#define DH      64
#define QKVO    512

// ---------------------------------------------------------------------------
// Kernel 1: fold BN into weights/biases (and attention scale into Q rows)
// ---------------------------------------------------------------------------
__global__ void k_fold(const float* __restrict__ qkv_w, const float* __restrict__ qkv_g,
                       const float* __restrict__ qkv_b, const float* __restrict__ qkv_m,
                       const float* __restrict__ qkv_v,
                       const float* __restrict__ c1_w, const float* __restrict__ c1_g,
                       const float* __restrict__ c1_b, const float* __restrict__ c1_m,
                       const float* __restrict__ c1_v,
                       const float* __restrict__ c2_w, const float* __restrict__ c2_g,
                       const float* __restrict__ c2_b, const float* __restrict__ c2_m,
                       const float* __restrict__ c2_v,
                       __bf16* __restrict__ wq, float* __restrict__ qbias,
                       float* __restrict__ w1f, float* __restrict__ b1f,
                       __bf16* __restrict__ w2f, float* __restrict__ b2f) {
  int t = threadIdx.x;                       // 512 threads
  float s = qkv_g[t] * rsqrtf(qkv_v[t] + 1e-3f);
  float bias = qkv_b[t] - qkv_m[t] * s;
  if ((t & 127) < DK) {                      // q rows: fold 1/sqrt(dim_key)
    const float att = 0.17677669529663687f;  // 32^-0.5
    s *= att; bias *= att;
  }
  qbias[t] = bias;
  for (int ic = 0; ic < CH; ++ic) wq[t * CH + ic] = f2bf(qkv_w[t * CH + ic] * s);
  if (t < CH) {
    float s1 = c1_g[t] * rsqrtf(c1_v[t] + 1e-3f);
    b1f[t] = c1_b[t] - c1_m[t] * s1;
    for (int k = 0; k < 9; ++k) w1f[t * 9 + k] = c1_w[t * 9 + k] * s1;
    float s2 = c2_g[t] * rsqrtf(c2_v[t] + 1e-3f);
    b2f[t] = c2_b[t] - c2_m[t] * s2;
    for (int ic = 0; ic < CH; ++ic) w2f[t * CH + ic] = f2bf(c2_w[t * CH + ic] * s2);
  }
}

// ---------------------------------------------------------------------------
// Kernel 2: x (NCHW f32) -> xt (token-major bf16 [tok][256]) via LDS transpose
// grid: 8 * (2304/32) blocks, 256 threads
// ---------------------------------------------------------------------------
__global__ void k_xt(const float* __restrict__ x, __bf16* __restrict__ xt) {
  __shared__ __bf16 tile[CH][33];     // [c][p], padded rows
  int blk = blockIdx.x;
  int b = blk / (HW / 32);
  int p0 = (blk % (HW / 32)) * 32;
  int t = threadIdx.x;
  int cl = t >> 5, pl = t & 31;
#pragma unroll
  for (int pass = 0; pass < 32; ++pass) {   // coalesced fp32 reads
    int c = pass * 8 + cl;
    tile[c][pl] = f2bf(x[((size_t)b * CH + c) * HW + p0 + pl]);
  }
  __syncthreads();
  int chunk = t & 31;                        // 16B chunk = 8 channels
#pragma unroll
  for (int pass = 0; pass < 4; ++pass) {     // coalesced b128 writes
    int row = pass * 8 + (t >> 5);
    union { v4u u; __bf16 h[8]; } pk;
#pragma unroll
    for (int q = 0; q < 8; ++q) pk.h[q] = tile[chunk * 8 + q][row];
    *(v4u*)&xt[((size_t)b * HW + p0 + row) * CH + chunk * 8] = pk.u;
  }
}

// ---------------------------------------------------------------------------
// Kernel 3: qkv GEMM (512 x 256 x 18432). Wave computes 64 oc x 16 tokens:
// B fragment reused across 4 WMMAs per K-step. Scatter to Qt/Kt/Vt.
//   Qt/Kt: [b*4+head][tok][32] bf16   Vt: [b][tok][256] bf16
// ---------------------------------------------------------------------------
__global__ void k_qkv_gemm(const __bf16* __restrict__ xt, const __bf16* __restrict__ wq,
                           const float* __restrict__ qbias,
                           __bf16* __restrict__ Qt, __bf16* __restrict__ Kt,
                           __bf16* __restrict__ Vt) {
  int lane = threadIdx.x & 31;
  int wv = blockIdx.x * 8 + (threadIdx.x >> 5);
  int og = wv & 7;                   // 8 groups of 64 oc
  int toktile = wv >> 3;             // 1152 token tiles
  int half = lane >> 4, l15 = lane & 15;
  const __bf16* a0 = wq + (size_t)(og * 64 + l15) * CH + half * 8;
  const __bf16* brow = xt + (size_t)(toktile * 16 + l15) * CH + half * 8;
  v8f acc[4] = {{}, {}, {}, {}};
#pragma unroll
  for (int k0 = 0; k0 < CH; k0 += 32) {
    v16bf b = load_frag(brow + k0, brow + k0 + 16);
    v16bf f0 = load_frag(a0 + k0, a0 + k0 + 16);
    v16bf f1 = load_frag(a0 + (size_t)16 * CH + k0, a0 + (size_t)16 * CH + k0 + 16);
    v16bf f2 = load_frag(a0 + (size_t)32 * CH + k0, a0 + (size_t)32 * CH + k0 + 16);
    v16bf f3 = load_frag(a0 + (size_t)48 * CH + k0, a0 + (size_t)48 * CH + k0 + 16);
    acc[0] = WMMA_BF16(f0, b, acc[0]);
    acc[1] = WMMA_BF16(f1, b, acc[1]);
    acc[2] = WMMA_BF16(f2, b, acc[2]);
    acc[3] = WMMA_BF16(f3, b, acc[3]);
  }
  int token = toktile * 16 + l15;
  int bb = token / HW, p = token % HW;
#pragma unroll
  for (int sub = 0; sub < 4; ++sub) {
#pragma unroll
    for (int r = 0; r < 8; ++r) {
      int oc = og * 64 + sub * 16 + r + 8 * half;
      float val = acc[sub][r] + qbias[oc];
      int hc = oc & 127, head = oc >> 7;
      if (hc < DK)
        Qt[((size_t)(bb * NHEAD + head) * HW + p) * DK + hc] = f2bf(val);
      else if (hc < 2 * DK)
        Kt[((size_t)(bb * NHEAD + head) * HW + p) * DK + (hc - DK)] = f2bf(val);
      else
        Vt[((size_t)bb * HW + p) * CH + head * DH + (hc - 2 * DK)] = f2bf(val);
    }
  }
}

// ---------------------------------------------------------------------------
// Kernel 4: flash attention. 4 waves/WG; wave owns a 16-row query tile.
//   attn_out: token-major [b*HW + p][256] fp32
// ---------------------------------------------------------------------------
__global__ void __launch_bounds__(128) k_attn(const __bf16* __restrict__ Qt,
                                              const __bf16* __restrict__ Kt,
                                              const __bf16* __restrict__ Vt,
                                              float* __restrict__ attn_out) {
  __shared__ __bf16 Ksh[32][32];        // [j][k]
  __shared__ __bf16 Vsh[64][32];        // [d][j]  (transposed during staging)
  __shared__ __bf16 Psh[4][16][32];     // per wave: [i][j]

  int t = threadIdx.x;
  int lane = t & 31, wv = t >> 5;
  int half = lane >> 4, l15 = lane & 15;
  int bh = blockIdx.y;                  // b*4 + head
  int b = bh >> 2, head = bh & 3;
  int i0 = blockIdx.x * 64 + wv * 16;

  const __bf16* kbase = Kt + (size_t)bh * HW * DK;
  const __bf16* vbase = Vt + (size_t)b * HW * CH + head * DH;

  const __bf16* qrow = Qt + (size_t)bh * HW * DK + (size_t)(i0 + l15) * DK;
  v16bf aq = load_frag(qrow + half * 8, qrow + 16 + half * 8);

  v8f acc0 = {}, acc1 = {}, acc2 = {}, acc3 = {};
  float m[8], l[8];
#pragma unroll
  for (int r = 0; r < 8; ++r) { m[r] = -1e30f; l[r] = 0.f; }

  int srow = t >> 2, sseg = t & 3;      // staging coords (32 rows x 4 segs)

  for (int j0 = 0; j0 < HW; j0 += 32) {
    __syncthreads();
    {   // cooperative staging: K chunk (32x32) + V chunk (32x64, transposed)
      *(v4u*)&Ksh[srow][sseg * 8] =
          *(const v4u*)(kbase + (size_t)(j0 + srow) * DK + sseg * 8);
      Frag f;
      const __bf16* vg = vbase + (size_t)(j0 + srow) * CH + sseg * 16;
      f.u[0] = *(const v4u*)vg;
      f.u[1] = *(const v4u*)(vg + 8);
#pragma unroll
      for (int q = 0; q < 16; ++q) Vsh[sseg * 16 + q][srow] = f.h[q];
      if (j0 + 32 < HW) {   // prefetch next chunk (global_prefetch_b8)
        __builtin_prefetch((const void*)(kbase + (size_t)(j0 + 32 + srow) * DK + sseg * 8), 0, 3);
        __builtin_prefetch((const void*)(vbase + (size_t)(j0 + 32 + srow) * CH + sseg * 16), 0, 3);
      }
    }
    __syncthreads();

    // S(16 x 32) = Q_i^T x K_j : two WMMAs (K contraction = dim_key = 32)
    const __bf16* k0r = &Ksh[l15][0];
    const __bf16* k1r = &Ksh[16 + l15][0];
    v16bf bk0 = load_frag(k0r + half * 8, k0r + 16 + half * 8);
    v16bf bk1 = load_frag(k1r + half * 8, k1r + 16 + half * 8);
    v8f s0 = {}, s1 = {};
    s0 = WMMA_BF16(aq, bk0, s0);
    s1 = WMMA_BF16(aq, bk1, s1);

    // Online softmax per row (row i = r + 8*half; 16 lanes per half hold its cols)
#pragma unroll
    for (int r = 0; r < 8; ++r) {
      float a = s0[r], c = s1[r];
      float mx = fmaxf(a, c);
#pragma unroll
      for (int off = 1; off < 16; off <<= 1) mx = fmaxf(mx, __shfl_xor(mx, off, 32));
      float mnew = fmaxf(m[r], mx);
      float alpha = __expf(m[r] - mnew);
      float p0 = __expf(a - mnew);
      float p1 = __expf(c - mnew);
      float rs = p0 + p1;
#pragma unroll
      for (int off = 1; off < 16; off <<= 1) rs += __shfl_xor(rs, off, 32);
      l[r] = l[r] * alpha + rs;
      m[r] = mnew;
      acc0[r] *= alpha; acc1[r] *= alpha; acc2[r] *= alpha; acc3[r] *= alpha;
      int mrow = r + 8 * half;
      Psh[wv][mrow][l15] = f2bf(p0);
      Psh[wv][mrow][16 + l15] = f2bf(p1);
    }

    // O(16 x 64) += P(16 x 32) x V^T(32 x 64): four WMMAs
    const __bf16* prow = &Psh[wv][l15][0];
    v16bf ap = load_frag(prow + half * 8, prow + 16 + half * 8);
    const __bf16* v0r = &Vsh[0 * 16 + l15][0];
    const __bf16* v1r = &Vsh[1 * 16 + l15][0];
    const __bf16* v2r = &Vsh[2 * 16 + l15][0];
    const __bf16* v3r = &Vsh[3 * 16 + l15][0];
    v16bf bv0 = load_frag(v0r + half * 8, v0r + 16 + half * 8);
    v16bf bv1 = load_frag(v1r + half * 8, v1r + 16 + half * 8);
    v16bf bv2 = load_frag(v2r + half * 8, v2r + 16 + half * 8);
    v16bf bv3 = load_frag(v3r + half * 8, v3r + 16 + half * 8);
    acc0 = WMMA_BF16(ap, bv0, acc0);
    acc1 = WMMA_BF16(ap, bv1, acc1);
    acc2 = WMMA_BF16(ap, bv2, acc2);
    acc3 = WMMA_BF16(ap, bv3, acc3);
  }

  // finalize: divide by row sums; token-major writes (contiguous per half-wave)
  float* obase = attn_out + ((size_t)(b * HW + i0)) * CH + head * DH;
#pragma unroll
  for (int r = 0; r < 8; ++r) {
    float inv = 1.f / l[r];
    size_t rowoff = (size_t)(r + 8 * half) * CH;
    obase[rowoff + 0 * 16 + l15] = acc0[r] * inv;
    obase[rowoff + 1 * 16 + l15] = acc1[r] * inv;
    obase[rowoff + 2 * 16 + l15] = acc2[r] * inv;
    obase[rowoff + 3 * 16 + l15] = acc3[r] * inv;
  }
}

// ---------------------------------------------------------------------------
// Kernel 5: y = attn_out + BN(depthwise3x3(V_img)) -> yt (token-major bf16)
// ---------------------------------------------------------------------------
__global__ void k_dw(const __bf16* __restrict__ Vt, const float* __restrict__ attn_tm,
                     const float* __restrict__ w1f, const float* __restrict__ b1f,
                     __bf16* __restrict__ yt) {
  int token = blockIdx.x;
  int b = token / HW, p = token % HW;
  int hh = p / HHW, ww = p % HHW;
  int c = threadIdx.x;
  float acc = b1f[c];
#pragma unroll
  for (int dy = -1; dy <= 1; ++dy) {
    int y = hh + dy;
    if ((unsigned)y >= (unsigned)HHW) continue;
#pragma unroll
    for (int dx = -1; dx <= 1; ++dx) {
      int xx = ww + dx;
      if ((unsigned)xx >= (unsigned)HHW) continue;
      float vv = bf2f(Vt[((size_t)b * HW + y * HHW + xx) * CH + c]);
      acc += w1f[c * 9 + (dy + 1) * 3 + (dx + 1)] * vv;
    }
  }
  acc += attn_tm[(size_t)token * CH + c];
  yt[(size_t)token * CH + c] = f2bf(acc);
}

// ---------------------------------------------------------------------------
// Kernel 6: final GEMM (256 x 256 x 18432). Wave computes 64 oc x 16 tokens.
// Output fp32 NCHW + bias.
// ---------------------------------------------------------------------------
__global__ void k_out_gemm(const __bf16* __restrict__ yt, const __bf16* __restrict__ w2f,
                           const float* __restrict__ b2f, float* __restrict__ out) {
  int lane = threadIdx.x & 31;
  int wv = blockIdx.x * 8 + (threadIdx.x >> 5);
  int og = wv & 3;                   // 4 groups of 64 oc
  int toktile = wv >> 2;             // 1152 token tiles
  int half = lane >> 4, l15 = lane & 15;
  const __bf16* a0 = w2f + (size_t)(og * 64 + l15) * CH + half * 8;
  const __bf16* brow = yt + (size_t)(toktile * 16 + l15) * CH + half * 8;
  v8f acc[4] = {{}, {}, {}, {}};
#pragma unroll
  for (int k0 = 0; k0 < CH; k0 += 32) {
    v16bf b = load_frag(brow + k0, brow + k0 + 16);
    v16bf f0 = load_frag(a0 + k0, a0 + k0 + 16);
    v16bf f1 = load_frag(a0 + (size_t)16 * CH + k0, a0 + (size_t)16 * CH + k0 + 16);
    v16bf f2 = load_frag(a0 + (size_t)32 * CH + k0, a0 + (size_t)32 * CH + k0 + 16);
    v16bf f3 = load_frag(a0 + (size_t)48 * CH + k0, a0 + (size_t)48 * CH + k0 + 16);
    acc[0] = WMMA_BF16(f0, b, acc[0]);
    acc[1] = WMMA_BF16(f1, b, acc[1]);
    acc[2] = WMMA_BF16(f2, b, acc[2]);
    acc[3] = WMMA_BF16(f3, b, acc[3]);
  }
  int token = toktile * 16 + l15;
  int b = token / HW, p = token % HW;
#pragma unroll
  for (int sub = 0; sub < 4; ++sub) {
#pragma unroll
    for (int r = 0; r < 8; ++r) {
      int oc = og * 64 + sub * 16 + r + 8 * half;
      out[((size_t)b * CH + oc) * HW + p] = acc[sub][r] + b2f[oc];
    }
  }
}

// ---------------------------------------------------------------------------
extern "C" void kernel_launch(void* const* d_in, const int* in_sizes, int n_in,
                              void* d_out, int out_size, void* d_ws, size_t ws_size,
                              hipStream_t stream) {
  const float* x     = (const float*)d_in[0];
  const float* qkv_w = (const float*)d_in[1];
  const float* qkv_g = (const float*)d_in[2];
  const float* qkv_b = (const float*)d_in[3];
  const float* qkv_m = (const float*)d_in[4];
  const float* qkv_v = (const float*)d_in[5];
  const float* c1_w  = (const float*)d_in[6];
  const float* c1_g  = (const float*)d_in[7];
  const float* c1_b  = (const float*)d_in[8];
  const float* c1_m  = (const float*)d_in[9];
  const float* c1_v  = (const float*)d_in[10];
  const float* c2_w  = (const float*)d_in[11];
  const float* c2_g  = (const float*)d_in[12];
  const float* c2_b  = (const float*)d_in[13];
  const float* c2_m  = (const float*)d_in[14];
  const float* c2_v  = (const float*)d_in[15];

  // carve workspace (256B aligned)
  size_t off = 0;
  char* ws = (char*)d_ws;
  auto carve = [&](size_t bytes) -> void* {
    void* p = ws + off;
    off += (bytes + 255) & ~(size_t)255;
    return p;
  };
  __bf16* wq    = (__bf16*)carve((size_t)QKVO * CH * 2);
  float*  qbias = (float*) carve((size_t)QKVO * 4);
  float*  w1f   = (float*) carve((size_t)CH * 9 * 4);
  float*  b1f   = (float*) carve((size_t)CH * 4);
  __bf16* w2f   = (__bf16*)carve((size_t)CH * CH * 2);
  float*  b2f   = (float*) carve((size_t)CH * 4);
  __bf16* xt    = (__bf16*)carve((size_t)NTOK * CH * 2);
  __bf16* Qt    = (__bf16*)carve((size_t)BATCH * NHEAD * HW * DK * 2);
  __bf16* Kt    = (__bf16*)carve((size_t)BATCH * NHEAD * HW * DK * 2);
  __bf16* Vt    = (__bf16*)carve((size_t)NTOK * CH * 2);
  float*  attn  = (float*) carve((size_t)NTOK * CH * 4);
  __bf16* yt    = (__bf16*)carve((size_t)NTOK * CH * 2);
  (void)ws_size; (void)in_sizes; (void)n_in; (void)out_size;

  k_fold<<<1, 512, 0, stream>>>(qkv_w, qkv_g, qkv_b, qkv_m, qkv_v,
                                c1_w, c1_g, c1_b, c1_m, c1_v,
                                c2_w, c2_g, c2_b, c2_m, c2_v,
                                wq, qbias, w1f, b1f, w2f, b2f);
  k_xt<<<BATCH * (HW / 32), 256, 0, stream>>>(x, xt);
  k_qkv_gemm<<<(8 * (NTOK / 16)) / 8, 256, 0, stream>>>(xt, wq, qbias, Qt, Kt, Vt);
  k_attn<<<dim3(HW / 64, BATCH * NHEAD), 128, 0, stream>>>(Qt, Kt, Vt, attn);
  k_dw<<<NTOK, 256, 0, stream>>>(Vt, attn, w1f, b1f, yt);
  k_out_gemm<<<(4 * (NTOK / 16)) / 8, 256, 0, stream>>>(yt, w2f, b2f, (float*)d_out);
}
